// Attention_56530359550323
// MI455X (gfx1250) — compile-verified, matrix operand order checked by reading, not run on hardware
//
#include <hip/hip_runtime.h>
#include <hip/hip_bf16.h>

// Problem constants (match reference)
#define B_      4
#define S_      2048
#define D_      4096
#define H_      4
#define HD_     128
#define QKV_N   768     // 512 (q) + 128 (k) + 128 (v)
#define ATTN_N  512     // H_*HD_

typedef __bf16 bf16_t;
typedef __attribute__((ext_vector_type(16))) __bf16 v16bf;
typedef __attribute__((ext_vector_type(8)))  float  v8f;

union Frag16 {
    v16bf  v;
    uint4  q[2];
    bf16_t e[16];
};

__device__ inline bf16_t f2bf(float f) {
    unsigned u = __float_as_uint(f);
    unsigned r = (u + 0x7fffu + ((u >> 16) & 1u)) >> 16;   // round-to-nearest-even
    unsigned short us = (unsigned short)r;
    bf16_t b;
    __builtin_memcpy(&b, &us, 2);
    return b;
}

// Fill a 16-bit A/B fragment (documented 16x32 layout) from fp32 memory.
__device__ inline void load_frag_f32(Frag16& a, const float* __restrict__ p) {
    float4 f0 = *(const float4*)(p);
    float4 f1 = *(const float4*)(p + 4);
    float4 f2 = *(const float4*)(p + 16);
    float4 f3 = *(const float4*)(p + 20);
    float t[16] = { f0.x, f0.y, f0.z, f0.w, f1.x, f1.y, f1.z, f1.w,
                    f2.x, f2.y, f2.z, f2.w, f3.x, f3.y, f3.z, f3.w };
#pragma unroll
    for (int i = 0; i < 16; ++i) a.e[i] = f2bf(t[i]);
}

// ---------------------------------------------------------------------------
// Weight prep: build bf16 transposed weights so GEMM B-fragments are
// contiguous per-lane 16B loads (lane = output column, K contiguous).
// ---------------------------------------------------------------------------
__global__ void build_wqkvT(const float* __restrict__ wq,
                            const float* __restrict__ wk,
                            const float* __restrict__ wv,
                            bf16_t* __restrict__ out /* [768][4096] */) {
    int idx = blockIdx.x * blockDim.x + threadIdx.x;
    if (idx >= QKV_N * D_) return;
    int n = idx / D_;
    int k = idx - n * D_;
    float v;
    if (n < 512)      v = wq[(size_t)k * 512 + n];
    else if (n < 640) v = wk[(size_t)k * 128 + (n - 512)];
    else              v = wv[(size_t)k * 128 + (n - 640)];
    out[idx] = f2bf(v);
}

__global__ void build_woT(const float* __restrict__ wo,
                          bf16_t* __restrict__ out /* [4096][512] */) {
    int idx = blockIdx.x * blockDim.x + threadIdx.x;
    if (idx >= D_ * ATTN_N) return;
    int n = idx / ATTN_N;           // output feature
    int k = idx - n * ATTN_N;       // reduction dim
    out[idx] = f2bf(wo[(size_t)k * D_ + n]);
}

// ---------------------------------------------------------------------------
// Generic WMMA GEMM: C[M,N] = A[M,K] * BT[N,K]^T
// block = 128 threads (4 waves). Wave tile = 32x64: two 16-row A strips share
// every B fragment (2 consecutive WMMAs per B load -> ~21 flop/byte from
// cache). Block tile = 128x64. A is fp32 (converted on the fly) or bf16
// row-major; BT bf16 row-major.
// ---------------------------------------------------------------------------
template<bool A_IS_F32>
__global__ __launch_bounds__(128)
void gemm_wmma(const void* __restrict__ Av, const bf16_t* __restrict__ BT,
               float* __restrict__ C, int M, int N, int K) {
    (void)M;
    const int lane = threadIdx.x & 31;
    const int wave = threadIdx.x >> 5;
    const int row0 = blockIdx.y * 128 + wave * 32;   // two 16-row strips
    const int col0 = blockIdx.x * 64;
    const int ml   = lane & 15;
    const int hl   = lane >> 4;          // K-half select per the 16-bit A layout

    v8f acc[2][4];
#pragma unroll
    for (int s = 0; s < 2; ++s)
#pragma unroll
        for (int c = 0; c < 4; ++c)
#pragma unroll
            for (int r = 0; r < 8; ++r) acc[s][c][r] = 0.f;

    for (int kb = 0; kb < K; kb += 32) {
        Frag16 a[2];
        if (A_IS_F32) {
            const float* A = (const float*)Av;
            const float* p0 = A + (size_t)(row0 + ml) * K + kb + hl * 8;
            const float* p1 = A + (size_t)(row0 + 16 + ml) * K + kb + hl * 8;
            if (kb + 32 < K) {
                __builtin_prefetch(p0 + 32, 0, 1);
                __builtin_prefetch(p1 + 32, 0, 1);
            }
            load_frag_f32(a[0], p0);
            load_frag_f32(a[1], p1);
        } else {
            const bf16_t* A = (const bf16_t*)Av;
            const bf16_t* p0 = A + (size_t)(row0 + ml) * K + kb + hl * 8;
            const bf16_t* p1 = A + (size_t)(row0 + 16 + ml) * K + kb + hl * 8;
            if (kb + 32 < K) {
                __builtin_prefetch(p0 + 32, 0, 1);
                __builtin_prefetch(p1 + 32, 0, 1);
            }
            a[0].q[0] = *(const uint4*)(p0);
            a[0].q[1] = *(const uint4*)(p0 + 16);
            a[1].q[0] = *(const uint4*)(p1);
            a[1].q[1] = *(const uint4*)(p1 + 16);
        }
#pragma unroll
        for (int c = 0; c < 4; ++c) {
            Frag16 bf;
            const bf16_t* p = BT + (size_t)(col0 + c * 16 + ml) * K + kb + hl * 8;
            bf.q[0] = *(const uint4*)(p);
            bf.q[1] = *(const uint4*)(p + 16);
            acc[0][c] = __builtin_amdgcn_wmma_f32_16x16x32_bf16(
                false, a[0].v, false, bf.v, (short)0, acc[0][c], false, false);
            acc[1][c] = __builtin_amdgcn_wmma_f32_16x16x32_bf16(
                false, a[1].v, false, bf.v, (short)0, acc[1][c], false, false);
        }
    }

#pragma unroll
    for (int s = 0; s < 2; ++s)
#pragma unroll
        for (int c = 0; c < 4; ++c)
#pragma unroll
            for (int r = 0; r < 8; ++r) {
                int row = row0 + s * 16 + r + 8 * hl;
                int col = col0 + c * 16 + ml;
                C[(size_t)row * N + col] = acc[s][c][r];
            }
}

// ---------------------------------------------------------------------------
// RoPE + per-head RMS-norm + attn position scaling.
// One block (4 waves) per token. Wave w handles Q head w; wave 0 also does K,
// wave 1 converts V (written transposed: (B,128,S)).
// ---------------------------------------------------------------------------
__global__ __launch_bounds__(128)
void rope_norm(const float* __restrict__ qkv, const int* __restrict__ positions,
               bf16_t* __restrict__ q_bf /* (B,H,S,128) */,
               bf16_t* __restrict__ k_bf /* (B,S,128)   */,
               bf16_t* __restrict__ vT   /* (B,128,S)   */) {
    const int t    = blockIdx.x;              // token = b*S + s
    const int b    = t / S_;
    const int s    = t - b * S_;
    const int lane = threadIdx.x & 31;
    const int wave = threadIdx.x >> 5;
    const float pf = (float)positions[s];

    // each lane handles rope pairs p0,p1 -> dims 4*lane .. 4*lane+3
    const int p0 = lane * 2, p1 = lane * 2 + 1;
    const float LOG_THETA = 13.1223633774f;   // ln(500000)
    float s0, c0, s1, c1;
    {
        float a0 = pf * __expf(-(float)p0 * (LOG_THETA / 64.f));
        float a1 = pf * __expf(-(float)p1 * (LOG_THETA / 64.f));
        __sincosf(a0, &s0, &c0);
        __sincosf(a1, &s1, &c1);
    }

    auto process = [&](const float* base, bf16_t* outp, float extra) {
        float x00 = base[2 * p0], x01 = base[2 * p0 + 1];
        float x10 = base[2 * p1], x11 = base[2 * p1 + 1];
        float o00 = x00 * c0 - x01 * s0;
        float o01 = x00 * s0 + x01 * c0;
        float o10 = x10 * c1 - x11 * s1;
        float o11 = x10 * s1 + x11 * c1;
        float ss = o00 * o00 + o01 * o01 + o10 * o10 + o11 * o11;
#pragma unroll
        for (int m = 1; m < 32; m <<= 1) ss += __shfl_xor(ss, m, 32);
        float scale = rsqrtf(ss * (1.f / 128.f) + 1e-6f) * extra;
        outp[2 * p0]     = f2bf(o00 * scale);
        outp[2 * p0 + 1] = f2bf(o01 * scale);
        outp[2 * p1]     = f2bf(o10 * scale);
        outp[2 * p1 + 1] = f2bf(o11 * scale);
    };

    // Q heads (attn_scale applied here; floor-scale term per reference)
    float attn_scale = logf(floorf((pf + 1.f) / 8192.f) + 1.f) * 0.1f + 1.f;
    process(qkv + (size_t)t * QKV_N + wave * HD_,
            q_bf + (((size_t)(b * H_ + wave)) * S_ + s) * HD_, attn_scale);

    if (wave == 0) {  // K head
        process(qkv + (size_t)t * QKV_N + 512,
                k_bf + ((size_t)b * S_ + s) * HD_, 1.f);
    }
    if (wave == 1) {  // V: convert + transpose
        const float* vb = qkv + (size_t)t * QKV_N + 640;
#pragma unroll
        for (int i = 0; i < 4; ++i) {
            int d = lane * 4 + i;
            vT[((size_t)b * HD_ + d) * S_ + s] = f2bf(vb[d]);
        }
    }
}

// ---------------------------------------------------------------------------
// Causal flash attention (GQA: 1 KV head shared by 4 Q heads).
// grid = B*H*(S/64); block = 128 threads (4 waves), wave = 16 query rows.
// Per 32-wide KV tile: 8 WMMA for Q*K^T, shuffle-reduced online softmax,
// P transposed through per-wave LDS (s_wait_dscnt), 8 WMMA for P*V.
// ---------------------------------------------------------------------------
__global__ __launch_bounds__(128)
void flash_attn(const bf16_t* __restrict__ q_bf, const bf16_t* __restrict__ k_bf,
                const bf16_t* __restrict__ vT, bf16_t* __restrict__ attn_out) {
    __shared__ bf16_t pbuf[4][16 * 32];   // per-wave P tile (16 rows x 32 cols)

    const int lane = threadIdx.x & 31;
    const int wave = threadIdx.x >> 5;
    const int ml   = lane & 15;
    const int hl   = lane >> 4;

    const int blk  = blockIdx.x;
    const int qblk = blk & 31;            // S/64 = 32
    const int h    = (blk >> 5) & 3;
    const int b    = blk >> 7;
    const int q0   = qblk * 64 + wave * 16;

    // Q fragments for the whole 16x128 strip
    Frag16 qf[4];
    {
        const bf16_t* qb =
            q_bf + (((size_t)(b * H_ + h)) * S_ + q0 + ml) * HD_ + hl * 8;
#pragma unroll
        for (int f = 0; f < 4; ++f) {
            qf[f].q[0] = *(const uint4*)(qb + f * 32);
            qf[f].q[1] = *(const uint4*)(qb + f * 32 + 16);
        }
    }

    v8f acc[8];
#pragma unroll
    for (int f = 0; f < 8; ++f)
#pragma unroll
        for (int r = 0; r < 8; ++r) acc[f][r] = 0.f;
    float mrow[8], lrow[8];
#pragma unroll
    for (int r = 0; r < 8; ++r) { mrow[r] = -1e30f; lrow[r] = 0.f; }

    const float sscale = 0.08838834764831845f;  // 1/sqrt(128)
    const bf16_t* kbase = k_bf + (size_t)b * S_ * HD_;
    const bf16_t* vbase = vT + (size_t)b * HD_ * S_;

    for (int kv0 = 0; kv0 < q0 + 16; kv0 += 32) {
        // ---- scores: 16 x 32 tile = two 16x16 C fragments ----
        v8f sc[2];
#pragma unroll
        for (int j = 0; j < 2; ++j) {
#pragma unroll
            for (int r = 0; r < 8; ++r) sc[j][r] = 0.f;
            const bf16_t* krow =
                kbase + (size_t)(kv0 + j * 16 + ml) * HD_ + hl * 8;
#pragma unroll
            for (int f = 0; f < 4; ++f) {
                Frag16 kf;
                kf.q[0] = *(const uint4*)(krow + f * 32);
                kf.q[1] = *(const uint4*)(krow + f * 32 + 16);
                sc[j] = __builtin_amdgcn_wmma_f32_16x16x32_bf16(
                    false, qf[f].v, false, kf.v, (short)0, sc[j], false, false);
            }
        }

        // ---- scale + causal mask + row max (16-lane half reductions) ----
        float tmax[8];
#pragma unroll
        for (int r = 0; r < 8; ++r) {
            int qg = q0 + r + 8 * hl;
#pragma unroll
            for (int j = 0; j < 2; ++j) {
                int kg  = kv0 + j * 16 + ml;
                float v = sc[j][r] * sscale;
                sc[j][r] = (kg > qg) ? -1e30f : v;
            }
            float tm = fmaxf(sc[0][r], sc[1][r]);
#pragma unroll
            for (int m = 1; m < 16; m <<= 1) tm = fmaxf(tm, __shfl_xor(tm, m, 32));
            tmax[r] = tm;
        }

        // ---- online softmax update ----
#pragma unroll
        for (int r = 0; r < 8; ++r) {
            float mn    = fmaxf(mrow[r], tmax[r]);
            float alpha = __expf(mrow[r] - mn);
            mrow[r]     = mn;
            float e0 = __expf(sc[0][r] - mn);
            float e1 = __expf(sc[1][r] - mn);
            sc[0][r] = e0; sc[1][r] = e1;
            float ts = e0 + e1;
#pragma unroll
            for (int m = 1; m < 16; m <<= 1) ts += __shfl_xor(ts, m, 32);
            lrow[r] = lrow[r] * alpha + ts;
#pragma unroll
            for (int f = 0; f < 8; ++f) acc[f][r] *= alpha;
        }

        // ---- transpose P (D-frag layout -> A-frag layout) through LDS ----
#pragma unroll
        for (int j = 0; j < 2; ++j)
#pragma unroll
            for (int r = 0; r < 8; ++r)
                pbuf[wave][(r + 8 * hl) * 32 + j * 16 + ml] = f2bf(sc[j][r]);
        asm volatile("s_wait_dscnt 0" ::: "memory");   // CDNA5 split DS counter
        Frag16 pa;
        pa.q[0] = *(const uint4*)(&pbuf[wave][ml * 32 + hl * 8]);
        pa.q[1] = *(const uint4*)(&pbuf[wave][ml * 32 + 16 + hl * 8]);

        // ---- P (16x32) * V (32x128): 8 WMMAs, B from transposed V ----
#pragma unroll
        for (int f = 0; f < 8; ++f) {
            Frag16 vf;
            const bf16_t* vrow =
                vbase + (size_t)(f * 16 + ml) * S_ + kv0 + hl * 8;
            vf.q[0] = *(const uint4*)(vrow);
            vf.q[1] = *(const uint4*)(vrow + 16);
            acc[f] = __builtin_amdgcn_wmma_f32_16x16x32_bf16(
                false, pa.v, false, vf.v, (short)0, acc[f], false, false);
        }
    }

    // ---- epilogue: normalize and store bf16 (feeds out-proj GEMM) ----
#pragma unroll
    for (int r = 0; r < 8; ++r) {
        float inv = 1.f / lrow[r];
        int row   = q0 + r + 8 * hl;
        bf16_t* orow = attn_out + ((size_t)b * S_ + row) * ATTN_N + h * HD_;
#pragma unroll
        for (int f = 0; f < 8; ++f)
            orow[f * 16 + ml] = f2bf(acc[f][r] * inv);
    }
}

// ---------------------------------------------------------------------------
extern "C" void kernel_launch(void* const* d_in, const int* in_sizes, int n_in,
                              void* d_out, int out_size, void* d_ws, size_t ws_size,
                              hipStream_t stream) {
    (void)in_sizes; (void)n_in; (void)out_size; (void)ws_size;
    const float* x   = (const float*)d_in[0];
    const int*   pos = (const int*)d_in[1];
    const float* wq  = (const float*)d_in[2];
    const float* wk  = (const float*)d_in[3];
    const float* wv  = (const float*)d_in[4];
    const float* wo  = (const float*)d_in[5];
    float* out = (float*)d_out;

    char* ws = (char*)d_ws;
    size_t off = 0;
    auto take = [&](size_t bytes) -> void* {
        void* p = ws + off;
        off += (bytes + 255) & ~(size_t)255;
        return p;
    };
    bf16_t* wqkvT  = (bf16_t*)take((size_t)QKV_N * D_ * 2);       //  6 MB
    bf16_t* woT    = (bf16_t*)take((size_t)D_ * ATTN_N * 2);      //  4 MB
    float*  qkv    = (float*) take((size_t)B_ * S_ * QKV_N * 4);  // 24 MB
    bf16_t* q_bf   = (bf16_t*)take((size_t)B_ * H_ * S_ * HD_ * 2);
    bf16_t* k_bf   = (bf16_t*)take((size_t)B_ * S_ * HD_ * 2);
    bf16_t* vTb    = (bf16_t*)take((size_t)B_ * HD_ * S_ * 2);
    bf16_t* attn_b = (bf16_t*)take((size_t)B_ * S_ * ATTN_N * 2);

    const int M = B_ * S_;  // 8192

    build_wqkvT<<<(QKV_N * D_ + 255) / 256, 256, 0, stream>>>(wq, wk, wv, wqkvT);
    build_woT<<<(D_ * ATTN_N + 255) / 256, 256, 0, stream>>>(wo, woT);

    // qkv = x @ [wq|wk|wv]   (M=8192, N=768, K=4096), fp32 A converted on load
    gemm_wmma<true><<<dim3(QKV_N / 64, M / 128), 128, 0, stream>>>(
        x, wqkvT, qkv, M, QKV_N, D_);

    rope_norm<<<B_ * S_, 128, 0, stream>>>(qkv, pos, q_bf, k_bf, vTb);

    flash_attn<<<B_ * H_ * (S_ / 64), 128, 0, stream>>>(q_bf, k_bf, vTb, attn_b);

    // out = attn @ wo        (M=8192, N=4096, K=512), bf16 A
    gemm_wmma<false><<<dim3(D_ / 64, M / 128), 128, 0, stream>>>(
        attn_b, woT, out, M, D_, ATTN_N);
}
